// YICAMultiHeadAttention_18906446037424
// MI455X (gfx1250) — matrix operand order repaired and checked by reading, not compiled, not run on hardware
//
#include <hip/hip_runtime.h>
#include <math.h>

#define HID   1024
#define NH    16
#define HD    64
#define BATCH 2
#define SEQ   2048

typedef __attribute__((ext_vector_type(16))) __bf16 v16bf;
typedef __attribute__((ext_vector_type(8)))  __bf16 v8bf;
typedef __attribute__((ext_vector_type(8)))  float  v8f;

__device__ __forceinline__ unsigned short f2bf(float f) {
  unsigned int u = __float_as_uint(f);
  u += 0x7FFFu + ((u >> 16) & 1u);            // round-to-nearest-even
  return (unsigned short)(u >> 16);
}

__device__ __forceinline__ v16bf cat16(v8bf lo, v8bf hi) {
  return __builtin_shufflevector(lo, hi, 0,1,2,3,4,5,6,7,8,9,10,11,12,13,14,15);
}

// A-matrix fragment (16x32 bf16): lane&15 = M row.
// low lanes hold K {0..7, 16..23}; high lanes hold K {8..15, 24..31}.
__device__ __forceinline__ v16bf frag_a(const unsigned short* rowp, int lane) {
  const int k0 = (lane & 16) ? 8 : 0;
  v8bf lo = *(const v8bf*)(rowp + k0);
  v8bf hi = *(const v8bf*)(rowp + k0 + 16);
  return cat16(lo, hi);
}

// B-matrix fragment (32x16 bf16): lane&15 = N col.
// low lanes hold K {0..7, 16..23}; high lanes hold K {8..15, 24..31}.
__device__ __forceinline__ v16bf frag_b(const unsigned short* colp, int lane) {
  const int k0 = (lane & 16) ? 8 : 0;
  v8bf lo = *(const v8bf*)(colp + k0);
  v8bf hi = *(const v8bf*)(colp + k0 + 16);
  return cat16(lo, hi);
}

// ---- DPP butterfly swaps within 16-lane rows (no LDS traffic) ----
// xor1 = quad_perm([1,0,3,2]) = 0xB1 ; xor2 = quad_perm([2,3,0,1]) = 0x4E
// xor4 = row_half_mirror = 0x141   ; xor8 = row_mirror = 0x140
template <int CTRL>
__device__ __forceinline__ float dpp_swap(float x) {
  int i = __float_as_int(x);
  int r = __builtin_amdgcn_update_dpp(i, i, CTRL, 0xF, 0xF, true);
  return __int_as_float(r);
}
__device__ __forceinline__ float row16_max(float x) {
  x = fmaxf(x, dpp_swap<0xB1>(x));
  x = fmaxf(x, dpp_swap<0x4E>(x));
  x = fmaxf(x, dpp_swap<0x141>(x));
  x = fmaxf(x, dpp_swap<0x140>(x));
  return x;
}
__device__ __forceinline__ float row16_sum(float x) {
  x += dpp_swap<0xB1>(x);
  x += dpp_swap<0x4E>(x);
  x += dpp_swap<0x141>(x);
  x += dpp_swap<0x140>(x);
  return x;
}

// ---------------- fp32 -> bf16 convert (4 elems/thread) ----------------
__global__ void cvt_f32_bf16(const float* __restrict__ in,
                             unsigned short* __restrict__ out, int n) {
  int i = (blockIdx.x * blockDim.x + threadIdx.x) * 4;
  if (i < n) {
    float4 f = *(const float4*)(in + i);
    unsigned int lo = (unsigned int)f2bf(f.x) | ((unsigned int)f2bf(f.y) << 16);
    unsigned int hi = (unsigned int)f2bf(f.z) | ((unsigned int)f2bf(f.w) << 16);
    *(uint2*)(out + i) = make_uint2(lo, hi);
  }
}

// ---------------- bf16 GEMM: C[M,N] = A[M,K] * W[N,K]^T ----------------
// 128x128 block tile, 256 threads = 8 waves (2 M x 4 N), wave tile 64x32.
template <bool OUT_F32>
__global__ __launch_bounds__(256)
void gemm_bf16(const unsigned short* __restrict__ A,
               const unsigned short* __restrict__ W,
               void* __restrict__ Cout, int M, int N, int K) {
  __shared__ __attribute__((aligned(16))) unsigned short As[128 * 40];
  __shared__ __attribute__((aligned(16))) unsigned short Bs[128 * 40];

  const int tid  = threadIdx.x;
  const int lane = tid & 31;
  const int wid  = tid >> 5;
  const int wm   = wid >> 2;              // 0..1
  const int wn   = wid & 3;               // 0..3
  const int m0   = blockIdx.x * 128;
  const int n0   = blockIdx.y * 128;

  v8f acc[4][2];
  const v8f vz = {0.f, 0.f, 0.f, 0.f, 0.f, 0.f, 0.f, 0.f};
#pragma unroll
  for (int i = 0; i < 4; ++i)
#pragma unroll
    for (int j = 0; j < 2; ++j) acc[i][j] = vz;

  for (int k0 = 0; k0 < K; k0 += 32) {
    __syncthreads();
    // stage 128x32 A tile and 128x32 W tile (16 B per thread per round)
#pragma unroll
    for (int r = 0; r < 2; ++r) {
      int chunk = tid + r * 256;
      int row   = chunk >> 2;
      int ko    = (chunk & 3) * 8;
      *(v8bf*)(&As[row * 40 + ko]) =
          *(const v8bf*)(&A[(size_t)(m0 + row) * K + k0 + ko]);
      *(v8bf*)(&Bs[row * 40 + ko]) =
          *(const v8bf*)(&W[(size_t)(n0 + row) * K + k0 + ko]);
    }
    if (k0 + 32 < K)  // speculative prefetch of next A tile line
      __builtin_prefetch(&A[(size_t)(m0 + (tid >> 1)) * K + k0 + 32], 0, 0);
    __syncthreads();

    v16bf bfr[2];
#pragma unroll
    for (int nt = 0; nt < 2; ++nt) {
      int col = wn * 32 + nt * 16 + (lane & 15);
      bfr[nt] = frag_b(&Bs[col * 40], lane);
    }
#pragma unroll
    for (int mt = 0; mt < 4; ++mt) {
      int row  = wm * 64 + mt * 16 + (lane & 15);
      v16bf afr = frag_a(&As[row * 40], lane);
#pragma unroll
      for (int nt = 0; nt < 2; ++nt) {
        acc[mt][nt] = __builtin_amdgcn_wmma_f32_16x16x32_bf16(
            false, afr, false, bfr[nt], (short)0, acc[mt][nt], false, false);
      }
    }
  }

  // epilogue: C layout -> lane&15 = N col, vgpr slot = M row (lanes>=16: +8)
#pragma unroll
  for (int mt = 0; mt < 4; ++mt) {
#pragma unroll
    for (int nt = 0; nt < 2; ++nt) {
      int col   = n0 + wn * 32 + nt * 16 + (lane & 15);
      int rbase = m0 + wm * 64 + mt * 16 + ((lane & 16) ? 8 : 0);
#pragma unroll
      for (int r = 0; r < 8; ++r) {
        float v = acc[mt][nt][r];
        if (OUT_F32)
          ((float*)Cout)[(size_t)(rbase + r) * N + col] = v;
        else
          ((unsigned short*)Cout)[(size_t)(rbase + r) * N + col] = f2bf(v);
      }
    }
  }
}

// ---------------- flash attention ----------------
// qkv: bf16 [B*S, 3*HID] (q|k|v per row). O: bf16 [B*S, HID].
// block = 128 threads (4 waves), handles 64 query rows of one (b, h).
__global__ __launch_bounds__(128)
void flash_attn(const unsigned short* __restrict__ qkv,
                const float* __restrict__ mask,
                unsigned short* __restrict__ O) {
  __shared__ __attribute__((aligned(16))) unsigned short Ks[64 * 72];     // [key][d]
  __shared__ __attribute__((aligned(16))) unsigned short Vs[64 * 72];     // [d][key] (transposed)
  __shared__ __attribute__((aligned(16))) unsigned short Ps[4][16 * 72];  // per-wave P

  const int tid  = threadIdx.x;
  const int lane = tid & 31;
  const int wid  = tid >> 5;
  const int bid  = blockIdx.x;
  const int qblk = bid % (SEQ / 64);
  const int h    = (bid / (SEQ / 64)) % NH;
  const int b    = bid / ((SEQ / 64) * NH);
  const int q0   = qblk * 64;
  const float scale = 0.125f;  // 1/sqrt(64)

  // Q fragments for this wave's 16 rows (2 K-steps over head_dim=64)
  const int qrow = q0 + wid * 16 + (lane & 15);
  const unsigned short* qptr =
      qkv + (size_t)(b * SEQ + qrow) * (3 * HID) + h * HD;
  v16bf qa0 = frag_a(qptr, lane);
  v16bf qa1 = frag_a(qptr + 32, lane);

  const v8f vz = {0.f, 0.f, 0.f, 0.f, 0.f, 0.f, 0.f, 0.f};
  v8f o[4];
  float rmax[8], rsum[8];
#pragma unroll
  for (int i = 0; i < 4; ++i) o[i] = vz;
#pragma unroll
  for (int r = 0; r < 8; ++r) { rmax[r] = -1e30f; rsum[r] = 0.f; }

  for (int kb = 0; kb < SEQ; kb += 64) {
    __syncthreads();
    // stage K block (row-major) and V block (transposed) into LDS
#pragma unroll
    for (int r = 0; r < 4; ++r) {
      int chunk = tid + r * 128;       // 0..511
      int key   = chunk >> 3;          // 0..63
      int doff  = (chunk & 7) * 8;     // 0..56
      const size_t rowbase = (size_t)(b * SEQ + kb + key) * (3 * HID);
      *(v8bf*)(&Ks[key * 72 + doff]) =
          *(const v8bf*)(&qkv[rowbase + HID + h * HD + doff]);
      union { v8bf v; unsigned short u[8]; } vt;
      vt.v = *(const v8bf*)(&qkv[rowbase + 2 * HID + h * HD + doff]);
#pragma unroll
      for (int j = 0; j < 8; ++j) Vs[(doff + j) * 72 + key] = vt.u[j];
    }
    __syncthreads();

    // S = (Q K^T) * scale + mask
    v8f st[4];
#pragma unroll
    for (int nt = 0; nt < 4; ++nt) {
      st[nt] = vz;
      int col = nt * 16 + (lane & 15);
      st[nt] = __builtin_amdgcn_wmma_f32_16x16x32_bf16(
          false, qa0, false, frag_b(&Ks[col * 72], lane), (short)0, st[nt],
          false, false);
      st[nt] = __builtin_amdgcn_wmma_f32_16x16x32_bf16(
          false, qa1, false, frag_b(&Ks[col * 72 + 32], lane), (short)0,
          st[nt], false, false);
    }
    const int rbase = q0 + wid * 16 + ((lane & 16) ? 8 : 0);
    const int cbase = kb + (lane & 15);
#pragma unroll
    for (int nt = 0; nt < 4; ++nt)
#pragma unroll
      for (int r = 0; r < 8; ++r)
        st[nt][r] = st[nt][r] * scale +
                    mask[((size_t)(b * SEQ) + rbase + r) * SEQ + cbase + nt * 16];

    // online softmax: rows live within 16-lane halves; DPP butterfly
    // reductions stay inside 16-lane rows -> pure VALU, no DS traffic.
#pragma unroll
    for (int r = 0; r < 8; ++r) {
      float m = fmaxf(fmaxf(st[0][r], st[1][r]), fmaxf(st[2][r], st[3][r]));
      m = row16_max(m);
      float nmax  = fmaxf(rmax[r], m);
      float alpha = __expf(rmax[r] - nmax);
      rmax[r] = nmax;
      float s = 0.f;
#pragma unroll
      for (int nt = 0; nt < 4; ++nt) {
        float p = __expf(st[nt][r] - nmax);
        st[nt][r] = p;
        s += p;
      }
      s = row16_sum(s);
      rsum[r] = rsum[r] * alpha + s;
#pragma unroll
      for (int nt = 0; nt < 4; ++nt) o[nt][r] *= alpha;
    }

    // round-trip P through LDS: C-layout -> A-layout (per-wave region,
    // same-wave DS ops are in-order, no barrier needed)
    {
      int prow = (lane & 16) ? 8 : 0;
      int pcol = lane & 15;
#pragma unroll
      for (int nt = 0; nt < 4; ++nt)
#pragma unroll
        for (int r = 0; r < 8; ++r)
          Ps[wid][(prow + r) * 72 + nt * 16 + pcol] = f2bf(st[nt][r]);
    }

    // O += P * V  (A = P 16x64, B = V keys x dims via transposed Vs)
#pragma unroll
    for (int ks = 0; ks < 2; ++ks) {
      v16bf pa = frag_a(&Ps[wid][(lane & 15) * 72 + ks * 32], lane);
#pragma unroll
      for (int nt = 0; nt < 4; ++nt) {
        v16bf bfr = frag_b(&Vs[(nt * 16 + (lane & 15)) * 72 + ks * 32], lane);
        o[nt] = __builtin_amdgcn_wmma_f32_16x16x32_bf16(
            false, pa, false, bfr, (short)0, o[nt], false, false);
      }
    }
  }

  // epilogue: normalize and store bf16 O[b, q, h*64 + d]
  const int rbase = q0 + wid * 16 + ((lane & 16) ? 8 : 0);
#pragma unroll
  for (int nt = 0; nt < 4; ++nt) {
    int d = nt * 16 + (lane & 15);
#pragma unroll
    for (int r = 0; r < 8; ++r) {
      float v = o[nt][r] / rsum[r];
      O[(size_t)(b * SEQ + rbase + r) * HID + h * HD + d] = f2bf(v);
    }
  }
}

// ---------------- host launcher ----------------
extern "C" void kernel_launch(void* const* d_in, const int* in_sizes, int n_in,
                              void* d_out, int out_size, void* d_ws,
                              size_t ws_size, hipStream_t stream) {
  const float* hidden = (const float*)d_in[0];   // [B,S,H]
  const float* mask   = (const float*)d_in[1];   // [B,1,S,S]
  const float* w_qkv  = (const float*)d_in[2];   // [3H,H]
  const float* w_o    = (const float*)d_in[3];   // [H,H]
  float* out = (float*)d_out;                    // [B,S,H] fp32

  char* ws = (char*)d_ws;
  unsigned short* Xb    = (unsigned short*)(ws);              //  8 MiB
  unsigned short* Wqkvb = (unsigned short*)(ws + 8388608);    //  6 MiB
  unsigned short* Wob   = (unsigned short*)(ws + 14680064);   //  2 MiB
  unsigned short* QKV   = (unsigned short*)(ws + 16777216);   // 24 MiB
  unsigned short* Ob    = (unsigned short*)(ws + 41943040);   //  8 MiB

  const int M  = BATCH * SEQ;       // 4096
  const int nX = M * HID;           // 4,194,304
  const int nWq = 3 * HID * HID;    // 3,145,728
  const int nWo = HID * HID;        // 1,048,576

  cvt_f32_bf16<<<(nX / 4 + 255) / 256, 256, 0, stream>>>(hidden, Xb, nX);
  cvt_f32_bf16<<<(nWq / 4 + 255) / 256, 256, 0, stream>>>(w_qkv, Wqkvb, nWq);
  cvt_f32_bf16<<<(nWo / 4 + 255) / 256, 256, 0, stream>>>(w_o, Wob, nWo);

  // QKV projection: [4096,1024] x [3072,1024]^T -> bf16 [4096,3072]
  gemm_bf16<false><<<dim3(M / 128, (3 * HID) / 128), 256, 0, stream>>>(
      Xb, Wqkvb, QKV, M, 3 * HID, HID);

  // attention: one block per (b, h, 64-query tile)
  flash_attn<<<BATCH * NH * (SEQ / 64), 128, 0, stream>>>(QKV, mask, Ob);

  // output projection: [4096,1024] x [1024,1024]^T -> fp32 out
  gemm_bf16<true><<<dim3(M / 128, HID / 128), 256, 0, stream>>>(
      Ob, Wob, out, M, HID, HID);
}
